// SampledSoftmaxPredictionHead_188978561129
// MI455X (gfx1250) — matrix operand order repaired and checked by reading, not compiled, not run on hardware
//
#include <hip/hip_runtime.h>
#include <hip/hip_bf16.h>
#include <cstdint>

// ---------------------------------------------------------------------------
// Sampled-softmax prediction head for MI455X (gfx1250, wave32, WMMA).
//
//   N = 131072 rows, D = 64, K = 1024 negatives, T = 1, EPS = 1e-10.
//
//   neg_logits = hidden @ e_neg^T   -> v_wmma_f32_16x16x32_f16 (f32 accum)
//   logsumexp over negatives uses a Cauchy-Schwarz upper bound instead of a
//   running max: 1 v_exp_f32 per element (v_exp competes with XDL WMMA for
//   the TRANS pipe, so halving exp count matters). Positive logit is f32
//   exact and merged into the logsumexp exactly at the end.
// ---------------------------------------------------------------------------

typedef __attribute__((ext_vector_type(16))) _Float16 v16h;
typedef __attribute__((ext_vector_type(8)))  float    v8f;

#define NROWS     131072
#define DDIM      64
#define KNEG      1024
#define NTILES    (KNEG / 16)     // 64 column tiles
#define EPSF      (1e-10f)

// A-fragment K index for 16-bit A 16x32 (ISA 7.12.2):
//   lanes 0-15 : VGPR0-3 -> K=0..7,  VGPR4-7 -> K=16..23
//   lanes16-31 : VGPR0-3 -> K=8..15, VGPR4-7 -> K=24..31
__device__ __forceinline__ int ka_of(int v, int hi) {
    return ((v < 4) ? (2 * v) : (2 * v + 8)) + hi * 8;
}
// B-fragment K index for 16-bit B 32x16 (same K order as A):
//   lanes 0-15 : VGPR0-3 -> K=0..7,  VGPR4-7 -> K=16..23
//   lanes16-31 : VGPR0-3 -> K=8..15, VGPR4-7 -> K=24..31
__device__ __forceinline__ int kb_of(int v, int hi) {
    return ((v < 4) ? (2 * v) : (2 * v + 8)) + hi * 8;
}

__device__ __forceinline__ uint32_t pack2h(float x, float y) {
    _Float16 a = (_Float16)x, b = (_Float16)y;
    unsigned short ua = __builtin_bit_cast(unsigned short, a);
    unsigned short ub = __builtin_bit_cast(unsigned short, b);
    return (uint32_t)ua | ((uint32_t)ub << 16);
}

// ---------------------------------------------------------------------------
// Kernel 1: pack e_neg^T (f16) directly into WMMA B-fragment layout so the
// GEMM loop does one coalesced 32B/lane load per fragment, plus
// lq[col] = log(sampling_probs[sampled[col]] + EPS).
// Bpack dword index = ((j*2+f)*32 + lane)*8 + v   (exactly == tid)
// ---------------------------------------------------------------------------
__global__ void ssph_prep_kernel(const int* __restrict__ sidx,
                                 const float* __restrict__ emb,
                                 const float* __restrict__ sprobs,
                                 uint32_t* __restrict__ Bpack,
                                 float* __restrict__ lq) {
    int tid = blockIdx.x * blockDim.x + threadIdx.x;   // 32768 threads
    if (tid < NTILES * 2 * 32 * 8) {
        int v  = tid & 7;
        int l  = (tid >> 3) & 31;
        int jf = tid >> 8;            // j*2 + f
        int f  = jf & 1;
        int j  = jf >> 1;
        int hi = l >> 4, lo = l & 15;
        int d   = f * 32 + kb_of(v, hi);   // inner-dim index (0..63)
        int col = j * 16 + lo;             // negative-sample column
        long long row = (long long)sidx[col] * DDIM;
        float x0 = emb[row + d];
        float x1 = emb[row + d + 1];
        Bpack[tid] = pack2h(x0, x1);
    }
    if (tid < KNEG) {
        lq[tid] = __logf(sprobs[sidx[tid]] + EPSF);
    }
}

// ---------------------------------------------------------------------------
// Kernel 2: scalars for the logit upper bound over the 1024 negatives:
//   bscal[0] = max_col ||e_neg[col]||   bscal[1] = min_col lq[col]
// ---------------------------------------------------------------------------
__global__ void __launch_bounds__(1024)
ssph_bound_kernel(const int* __restrict__ sidx,
                  const float* __restrict__ emb,
                  const float* __restrict__ lq,
                  float* __restrict__ bscal) {
    __shared__ float se[KNEG];
    __shared__ float sl[KNEG];
    int t = threadIdx.x;                       // one block of 1024 threads
    long long row = (long long)sidx[t] * DDIM;
    float n2 = 0.f;
    #pragma unroll
    for (int d = 0; d < DDIM; d += 4) {
        float4 e = *(const float4*)(emb + row + d);
        n2 += e.x * e.x + e.y * e.y + e.z * e.z + e.w * e.w;
    }
    se[t] = sqrtf(n2);
    sl[t] = lq[t];
    __syncthreads();
    for (int off = 512; off > 0; off >>= 1) {
        if (t < off) {
            se[t] = fmaxf(se[t], se[t + off]);
            sl[t] = fminf(sl[t], sl[t + off]);
        }
        __syncthreads();
    }
    if (t == 0) { bscal[0] = se[0]; bscal[1] = sl[0]; }
}

// ---------------------------------------------------------------------------
// Kernel 3: per-wave 16-row tile. Per column tile: 1 ds_load_b64 (lq|sid),
// 2 coalesced B loads, 2 v_wmma_f32_16x16x32_f16, 8 exp-accumulate updates.
// ---------------------------------------------------------------------------
__global__ void __launch_bounds__(256)
ssph_main_kernel(const float* __restrict__ hidden,
                 const int*   __restrict__ y,
                 const int*   __restrict__ sidx,
                 const float* __restrict__ emb,
                 const float* __restrict__ sprobs,
                 const uint32_t* __restrict__ Bpack,
                 const float* __restrict__ lq,
                 const float* __restrict__ bscal,
                 float* __restrict__ ce_out,
                 float* __restrict__ mk_out) {
    __shared__ float2 sq[KNEG];         // {lq[col], as_float(sidx[col])}
    __shared__ float  hn2s[8][16];      // per-wave row ||h||^2
    __shared__ float  reds[8][16];      // per-wave row sum-of-exp

    const int lane = threadIdx.x & 31;
    const int w    = threadIdx.x >> 5;            // wave in block (0..7)
    const int m0   = (blockIdx.x * 8 + w) * 16;   // 16-row tile base
    const int lo   = lane & 15;
    const int hi   = lane >> 4;

    // stage lq/sidx interleaved into LDS (one ds_load_b64 per tile later)
    for (int i = threadIdx.x; i < KNEG; i += 256) {
        sq[i] = make_float2(lq[i], __int_as_float(sidx[i]));
    }

    // ---- A fragments (hidden rows m0..m0+15 as f16) + row norms ------------
    const float2* hrow = (const float2*)(hidden + (size_t)(m0 + lo) * DDIM);
    v16h a0, a1;
    float hn2 = 0.f;
    #pragma unroll
    for (int v = 0; v < 8; ++v) {
        int k = ka_of(v, hi);
        float2 h0 = hrow[k >> 1];            // d = k      (frag 0: d 0..31)
        float2 h1 = hrow[(k + 32) >> 1];     // d = k + 32 (frag 1: d 32..63)
        a0[2 * v]     = (_Float16)h0.x;
        a0[2 * v + 1] = (_Float16)h0.y;
        a1[2 * v]     = (_Float16)h1.x;
        a1[2 * v + 1] = (_Float16)h1.y;
        hn2 += h0.x * h0.x + h0.y * h0.y + h1.x * h1.x + h1.y * h1.y;
    }
    hn2 += __shfl_xor(hn2, 16, 32);          // both d-halves of row lo
    if (lane < 16) hn2s[w][lo] = hn2;
    __syncthreads();

    const float maxE  = bscal[0];            // uniform -> scalar loads
    const float minLq = bscal[1];

    // upper bound for this lane's 8 C rows (C row = v + 8*hi)
    float bnd[8];
    #pragma unroll
    for (int v = 0; v < 8; ++v)
        bnd[v] = sqrtf(hn2s[w][8 * hi + v]) * maxE - minLq;

    int yv[8];
    #pragma unroll
    for (int v = 0; v < 8; ++v) yv[v] = y[m0 + 8 * hi + v];

    // ---- GEMM + bounded sum-of-exp over 64 column tiles --------------------
    float sm[8];
    #pragma unroll
    for (int v = 0; v < 8; ++v) sm[v] = 0.f;

    for (int j = 0; j < NTILES; ++j) {
        const uint32_t* bp = Bpack + (size_t)j * 512 + lane * 8;
        v16h b0 = *(const v16h*)(bp);
        v16h b1 = *(const v16h*)(bp + 256);
        if (j + 4 < NTILES)
            __builtin_prefetch(bp + 4 * 512, 0, 3);   // global_prefetch_b8
        v8f c = {};
        c = __builtin_amdgcn_wmma_f32_16x16x32_f16(false, a0, false, b0,
                                                   (short)0, c, false, false);
        c = __builtin_amdgcn_wmma_f32_16x16x32_f16(false, a1, false, b1,
                                                   (short)0, c, false, false);
        float2 p  = sq[j * 16 + lo];
        float lqv = p.x;
        int   sid = __float_as_int(p.y);
        #pragma unroll
        for (int v = 0; v < 8; ++v) {
            float e = __expf(c[v] - lqv - bnd[v]);    // arg <= 0 by bound
            e = (yv[v] == sid) ? 0.f : e;             // collision mask
            sm[v] += e;
        }
    }

    // ---- sum over the 16 lanes sharing each row (plain adds) ---------------
    #pragma unroll
    for (int v = 0; v < 8; ++v) {
        float s_ = sm[v];
        s_ += __shfl_xor(s_, 1, 32);
        s_ += __shfl_xor(s_, 2, 32);
        s_ += __shfl_xor(s_, 4, 32);
        s_ += __shfl_xor(s_, 8, 32);
        sm[v] = s_;
    }
    if (lo == 0) {                        // lanes 0 and 16
        #pragma unroll
        for (int v = 0; v < 8; ++v) reds[w][hi * 8 + v] = sm[v];
    }
    __syncthreads();
    float rs = reds[w][lo];               // row lo's neg sum at lane lo

    // ---- positive logit for row (m0+lo), exact f32 -------------------------
    int yrow = y[m0 + lo];
    float pacc = 0.f;
    long long ebase = (long long)yrow * DDIM;
    #pragma unroll
    for (int v = 0; v < 8; ++v) {
        int k = ka_of(v, hi);
        float2 h0 = hrow[k >> 1];
        float2 h1 = hrow[(k + 32) >> 1];
        float2 e0 = *(const float2*)(emb + ebase + k);
        float2 e1 = *(const float2*)(emb + ebase + k + 32);
        pacc += h0.x * e0.x + h0.y * e0.y + h1.x * e1.x + h1.y * e1.y;
    }
    pacc += __shfl_xor(pacc, 16, 32);     // combine the two d-halves
    float pos = pacc - __logf(sprobs[yrow] + EPSF);

    // ---- exact 2-way merge of (bnd, rs) with (pos, 1); ce = lse - pos ------
    if (lane < 16) {
        float bnd_r = sqrtf(hn2s[w][lane]) * maxE - minLq;  // == row's bnd
        float nm  = fmaxf(bnd_r, pos);
        float s   = rs * __expf(bnd_r - nm) + __expf(pos - nm);
        float lse = nm + __logf(s);
        float msk = (yrow != 0) ? 1.f : 0.f;
        int row = m0 + lane;
        ce_out[row] = (lse - pos) * msk;
        mk_out[row] = msk;
    }
}

// ---------------------------------------------------------------------------
// Deterministic fixed-shape reductions (no float atomics).
// ---------------------------------------------------------------------------
__global__ void ssph_reduce1_kernel(const float* __restrict__ ce,
                                    const float* __restrict__ mk,
                                    float* __restrict__ pc,
                                    float* __restrict__ pm) {
    __shared__ float sc[256], sq2[256];
    int t = threadIdx.x;
    int base = blockIdx.x * 512;
    sc[t]  = ce[base + t] + ce[base + 256 + t];
    sq2[t] = mk[base + t] + mk[base + 256 + t];
    __syncthreads();
    for (int off = 128; off > 0; off >>= 1) {
        if (t < off) { sc[t] += sc[t + off]; sq2[t] += sq2[t + off]; }
        __syncthreads();
    }
    if (t == 0) { pc[blockIdx.x] = sc[0]; pm[blockIdx.x] = sq2[0]; }
}

__global__ void ssph_reduce2_kernel(const float* __restrict__ pc,
                                    const float* __restrict__ pm,
                                    float* __restrict__ out) {
    __shared__ float sc[256], sq2[256];
    int t = threadIdx.x;
    sc[t]  = pc[t];
    sq2[t] = pm[t];
    __syncthreads();
    for (int off = 128; off > 0; off >>= 1) {
        if (t < off) { sc[t] += sc[t + off]; sq2[t] += sq2[t + off]; }
        __syncthreads();
    }
    if (t == 0) out[0] = sc[0] / fmaxf(sq2[0], 1.0f);
}

// ---------------------------------------------------------------------------
extern "C" void kernel_launch(void* const* d_in, const int* in_sizes, int n_in,
                              void* d_out, int out_size, void* d_ws, size_t ws_size,
                              hipStream_t stream) {
    const float* hidden = (const float*)d_in[0];   // [N, 64] f32
    const int*   y      = (const int*)  d_in[1];   // [N]
    const int*   sidx   = (const int*)  d_in[2];   // [1024]
    const float* emb    = (const float*)d_in[3];   // [1e6, 64] f32
    const float* sprobs = (const float*)d_in[4];   // [1e6] f32
    float*       out    = (float*)d_out;

    char* ws = (char*)d_ws;
    uint32_t* Bpack = (uint32_t*)ws;                              // 128 KiB
    float*    lq    = (float*)(ws + 131072);                      //   4 KiB
    float*    ce    = (float*)(ws + 131072 + 4096);               // 512 KiB
    float*    mk    = (float*)(ws + 131072 + 4096 + 524288);      // 512 KiB
    float*    pc    = (float*)(ws + 131072 + 4096 + 2 * 524288);  //   1 KiB
    float*    pm    = pc + 256;                                   //   1 KiB
    float*    bscal = pm + 256;                                   //   2 floats

    ssph_prep_kernel<<<128, 256, 0, stream>>>(sidx, emb, sprobs, Bpack, lq);
    ssph_bound_kernel<<<1, 1024, 0, stream>>>(sidx, emb, lq, bscal);
    ssph_main_kernel<<<NROWS / 16 / 8, 256, 0, stream>>>(
        hidden, y, sidx, emb, sprobs, Bpack, lq, bscal, ce, mk);
    ssph_reduce1_kernel<<<256, 256, 0, stream>>>(ce, mk, pc, pm);
    ssph_reduce2_kernel<<<1, 256, 0, stream>>>(pc, pm, out);
}
